// BlockwiseHadamardInputWrapper_17686675325360
// MI455X (gfx1250) — compile-verified, matrix operand order checked by reading, not run on hardware
//
#include <hip/hip_runtime.h>
#include <hip/hip_bf16.h>
#include <stdint.h>

// ---------- types ----------
typedef __attribute__((ext_vector_type(16))) __bf16 v16bf;
typedef __attribute__((ext_vector_type(8)))  float  v8f;

union FragAB {
    v16bf v;
    uint4 q[2];
};

__device__ __forceinline__ uint16_t f2bf(float f) {
    union { float f; uint32_t u; } c; c.f = f;
    uint32_t u = c.u;
    u += 0x7FFFu + ((u >> 16) & 1u);   // round-to-nearest-even
    return (uint16_t)(u >> 16);
}

// ---------- async global->LDS copy (16 B per lane), with sync fallback ----------
#if __has_builtin(__builtin_amdgcn_global_load_async_to_lds_b128)
#define HAVE_ASYNC_LDS 1
typedef int v4i_ __attribute__((__vector_size__(16)));          // int4 vector
typedef __attribute__((address_space(1))) v4i_ gv4i;            // global int4
typedef __attribute__((address_space(3))) v4i_ lv4i;            // LDS int4
__device__ __forceinline__ void async_cp16(const uint16_t* g, uint16_t* l) {
    __builtin_amdgcn_global_load_async_to_lds_b128((gv4i*)g, (lv4i*)l, 0, 0);
}
#else
#define HAVE_ASYNC_LDS 0
__device__ __forceinline__ void async_cp16(const uint16_t* g, uint16_t* l) {
    *reinterpret_cast<uint4*>(l) = *reinterpret_cast<const uint4*>(g);
}
#endif

__device__ __forceinline__ void wait_async0() {
#if HAVE_ASYNC_LDS
#if __has_builtin(__builtin_amdgcn_s_wait_asynccnt)
    __builtin_amdgcn_s_wait_asynccnt(0);
#else
    asm volatile("s_wait_asynccnt 0" ::: "memory");
#endif
#endif
}

// ---------- kernel 1: W' = blockHadamard_128(W rows), f32 -> bf16 ----------
__global__ void hadamard_w_kernel(const float* __restrict__ W,
                                  uint16_t* __restrict__ Wh,
                                  int D, int nblk) {
    __shared__ float s[128];
    const int t   = threadIdx.x;
    const int row = blockIdx.x / nblk;
    const int cb  = blockIdx.x % nblk;
    const size_t base = (size_t)row * D + (size_t)cb * 128;

    s[t] = W[base + t];
    #pragma unroll
    for (int st = 1; st < 128; st <<= 1) {
        __syncthreads();
        const float a = s[t];
        const float b = s[t ^ st];
        const float v = (t & st) ? (b - a) : (a + b);
        __syncthreads();
        s[t] = v;
    }
    __syncthreads();
    Wh[base + t] = f2bf(s[t] * 0.08838834764831845f);  // 1/sqrt(128)
}

// ---------- kernel 2: x f32 -> bf16, 8 elems/thread ----------
__global__ void cvt_x_kernel(const float* __restrict__ x,
                             uint16_t* __restrict__ xb, size_t n) {
    const size_t i = ((size_t)blockIdx.x * blockDim.x + threadIdx.x) * 8;
    if (i >= n) return;
    const float4 f0 = *reinterpret_cast<const float4*>(x + i);
    const float4 f1 = *reinterpret_cast<const float4*>(x + i + 4);
    uint16_t h[8];
    h[0] = f2bf(f0.x); h[1] = f2bf(f0.y); h[2] = f2bf(f0.z); h[3] = f2bf(f0.w);
    h[4] = f2bf(f1.x); h[5] = f2bf(f1.y); h[6] = f2bf(f1.z); h[7] = f2bf(f1.w);
    *reinterpret_cast<uint4*>(xb + i) = *reinterpret_cast<uint4*>(h);
}

// ---------- kernel 3: C = Xb @ Wh^T + bias  (bf16 WMMA, f32 accum) ----------
// Workgroup: 256 threads = 8 waves (wave32). Block tile 128(M) x 256(N), BK=32.
// Waves 2(M) x 4(N); each wave computes 64x64 = 4x4 WMMA tiles (128 accum VGPRs).
// Double-buffered LDS; tiles staged via async global->LDS DMA (ASYNCcnt),
// one s_wait_asynccnt + one barrier per K step.
#define BM 128
#define BN 256
#define BK 32
#define LDST 40   // LDS row stride in bf16 elems (32 data + 8 pad = 80 B rows)

__global__ __launch_bounds__(256)
void gemm_bf16_kernel(const uint16_t* __restrict__ Xb,
                      const uint16_t* __restrict__ Wh,
                      const float* __restrict__ bias,
                      float* __restrict__ out,
                      int M, int N, int K) {
    __shared__ __align__(16) uint16_t As[2][BM * LDST];   // 2 x 10 KB
    __shared__ __align__(16) uint16_t Bs[2][BN * LDST];   // 2 x 20 KB

    const int tid  = threadIdx.x;
    const int bm0  = blockIdx.y * BM;
    const int bn0  = blockIdx.x * BN;

    const int wave = tid >> 5;         // 0..7
    const int lane = tid & 31;
    const int wm   = wave & 1;         // 0..1  -> 64-row slab
    const int wn   = wave >> 1;        // 0..3  -> 64-col slab
    const int half = lane >> 4;        // 0/1 (K half-lane split)
    const int lrow = lane & 15;        // row (A) / col (B) within 16

    // global->LDS loader mapping
    const int lrA = tid >> 1;          // 0..127, A: half-row (16 elems) per thread
    const int lcA = (tid & 1) * 16;
    const int lrB = tid;               // 0..255, B: full 32-elem row per thread
    const uint16_t* gA = Xb + (size_t)(bm0 + lrA) * K + lcA;
    const uint16_t* gB = Wh + (size_t)(bn0 + lrB) * K;

    v8f acc[4][4];
    const v8f vzero = {0.f, 0.f, 0.f, 0.f, 0.f, 0.f, 0.f, 0.f};
    #pragma unroll
    for (int i = 0; i < 4; ++i)
        #pragma unroll
        for (int j = 0; j < 4; ++j)
            acc[i][j] = vzero;

    // staged-copy issue: 6 x 16B per thread (A: 2 chunks, B: 4 chunks)
    auto issue_tile = [&](int buf, int k) {
        uint16_t* la = &As[buf][lrA * LDST + lcA];
        uint16_t* lb = &Bs[buf][lrB * LDST];
        async_cp16(gA + k,      la);
        async_cp16(gA + k + 8,  la + 8);
        async_cp16(gB + k,      lb);
        async_cp16(gB + k + 8,  lb + 8);
        async_cp16(gB + k + 16, lb + 16);
        async_cp16(gB + k + 24, lb + 24);
    };

    // ---- prologue: stage tile k0=0 into buffer 0 ----
    issue_tile(0, 0);
    wait_async0();
    __syncthreads();

    int cur = 0;
    for (int k0 = 0; k0 < K; k0 += BK) {
        const bool more = (k0 + BK) < K;

        // kick off DMA of the next tile into the other buffer (runs under WMMA)
        if (more) {
            issue_tile(cur ^ 1, k0 + BK);
            __builtin_prefetch(gA + k0 + 2 * BK, 0, 1);   // global_prefetch_b8
            __builtin_prefetch(gB + k0 + 2 * BK, 0, 1);
        }

        // ---- compute: 16 x v_wmma_f32_16x16x32_bf16 from buf[cur] ----
        const int ko1 = half * 8;         // v16bf elems 0..7
        const int ko2 = 16 + half * 8;    // v16bf elems 8..15

        FragAB aF[4];
        #pragma unroll
        for (int i = 0; i < 4; ++i) {
            const int base = (wm * 64 + i * 16 + lrow) * LDST;
            aF[i].q[0] = *reinterpret_cast<const uint4*>(&As[cur][base + ko1]);
            aF[i].q[1] = *reinterpret_cast<const uint4*>(&As[cur][base + ko2]);
        }
        FragAB bF[4];
        #pragma unroll
        for (int j = 0; j < 4; ++j) {
            const int base = (wn * 64 + j * 16 + lrow) * LDST;
            bF[j].q[0] = *reinterpret_cast<const uint4*>(&Bs[cur][base + ko1]);
            bF[j].q[1] = *reinterpret_cast<const uint4*>(&Bs[cur][base + ko2]);
        }

        #pragma unroll
        for (int i = 0; i < 4; ++i)
            #pragma unroll
            for (int j = 0; j < 4; ++j)
                acc[i][j] = __builtin_amdgcn_wmma_f32_16x16x32_bf16(
                    /*neg_a=*/false, aF[i].v,
                    /*neg_b=*/false, bF[j].v,
                    /*c_mod=*/(short)0, acc[i][j],
                    /*reuse_a=*/false, /*reuse_b=*/false);

        // ensure this wave's DMA landed, then swap buffers workgroup-wide
        if (more) wait_async0();
        __syncthreads();
        cur ^= 1;
    }

    // ---- epilogue: bias + store ----
    float bj[4];
    #pragma unroll
    for (int j = 0; j < 4; ++j)
        bj[j] = bias[bn0 + wn * 64 + j * 16 + lrow];

    #pragma unroll
    for (int i = 0; i < 4; ++i) {
        #pragma unroll
        for (int j = 0; j < 4; ++j) {
            const int col = bn0 + wn * 64 + j * 16 + lrow;
            #pragma unroll
            for (int r = 0; r < 8; ++r) {
                const int row = bm0 + wm * 64 + i * 16 + half * 8 + r;
                out[(size_t)row * N + col] = acc[i][j][r] + bj[j];
            }
        }
    }
}

// ---------- launch ----------
extern "C" void kernel_launch(void* const* d_in, const int* in_sizes, int n_in,
                              void* d_out, int out_size, void* d_ws, size_t ws_size,
                              hipStream_t stream) {
    const float* x  = (const float*)d_in[0];   // [M, K] flat (B*S, D)
    const float* W  = (const float*)d_in[1];   // [N, K]
    const float* b  = (const float*)d_in[2];   // [N]

    const int N = in_sizes[2];                            // O
    const int K = in_sizes[1] / N;                        // D
    const int M = (int)((size_t)in_sizes[0] / (size_t)K); // B*S

    uint16_t* Xb = (uint16_t*)d_ws;                                // M*K bf16
    uint16_t* Wg = (uint16_t*)((char*)d_ws + (size_t)M * K * 2);   // N*K bf16
    float*    out = (float*)d_out;

    // 1) W' = W * H_d (blockwise FWHT on rows), cast bf16
    {
        const int nblk = K / 128;
        dim3 grid((unsigned)(N * nblk));
        hipLaunchKernelGGL(hadamard_w_kernel, grid, dim3(128), 0, stream, W, Wg, K, nblk);
    }
    // 2) x -> bf16
    {
        const size_t n = (size_t)M * K;
        const unsigned blocks = (unsigned)((n + 2047) / 2048);   // 256 thr * 8 elems
        hipLaunchKernelGGL(cvt_x_kernel, dim3(blocks), dim3(256), 0, stream, x, Xb, n);
    }
    // 3) GEMM: out = Xb @ Wg^T + b
    {
        dim3 grid((unsigned)(N / BN), (unsigned)(M / BM));
        hipLaunchKernelGGL(gemm_bf16_kernel, grid, dim3(256), 0, stream,
                           Xb, Wg, b, out, M, N, K);
    }
}
